// GRU_979252543774
// MI455X (gfx1250) — matrix operand order repaired and checked by reading, not compiled, not run on hardware
//
#include <hip/hip_runtime.h>
#include <hip/hip_bf16.h>
#include <math.h>

// Problem constants (from reference): L,B,S,H,E,V
#define L_ 2
#define B_ 64
#define S_ 64
#define H_ 512
#define E_ 512
#define V_ 32000

typedef __bf16 bf16;
typedef __attribute__((ext_vector_type(16))) __bf16 v16bf;
typedef __attribute__((ext_vector_type(8)))  __bf16 v8bf;
typedef __attribute__((ext_vector_type(8)))  float  v8f;

static __device__ inline v8f wmma_bf16(v16bf a, v16bf b, v8f c) {
  // (neg_a, A, neg_b, B, c_mod, C, reuse_a, reuse_b)
  return __builtin_amdgcn_wmma_f32_16x16x32_bf16(false, a, false, b, (short)0, c,
                                                 false, false);
}

static __device__ inline v8f zero8() {
  v8f z = {0.f, 0.f, 0.f, 0.f, 0.f, 0.f, 0.f, 0.f};
  return z;
}

static __device__ inline float sigm(float x) { return 1.0f / (1.0f + __expf(-x)); }

// A fragment: 16x32 bf16 tile at (row0, k0) of a row-major [rows x ld] bf16 matrix.
// Lane l: m = l&15; elems j=0..15 -> K = k0 + 16*(l>>4) + j (contiguous).
static __device__ inline v16bf load_a(const bf16* __restrict__ base, int ld,
                                      int row0, int k0, int lane) {
  const int r  = row0 + (lane & 15);
  const int kh = (lane >> 4) << 4;
  const bf16* p = base + (size_t)r * ld + k0 + kh;
  v8bf lo = *(const v8bf*)p;
  v8bf hi = *(const v8bf*)(p + 8);
  v16bf a;
#pragma unroll
  for (int i = 0; i < 8; ++i) { a[i] = lo[i]; a[i + 8] = hi[i]; }
  return a;
}

// B fragment for C = X * W^T: B[k][n] = W[n][k], W row-major [N x ld].
// Lane l: n = n0 + (l&15); elems j=0..15 -> K = k0 + 16*(l>>4) + j (contiguous).
// fp32 source: convert on load (fallback path).
static __device__ inline v16bf load_w16(const float* __restrict__ W, int ld,
                                        int n0, int k0, int lane) {
  const int n  = n0 + (lane & 15);
  const int kk = k0 + ((lane >> 4) << 4);
  const float* p = W + (size_t)n * ld + kk;
  v16bf b;
#pragma unroll
  for (int i = 0; i < 16; ++i) b[i] = (bf16)p[i];
  return b;
}

// bf16 source (weight-cache path): 16 contiguous bf16 = two b128 loads, no cvt.
static __device__ inline v16bf load_w16(const bf16* __restrict__ W, int ld,
                                        int n0, int k0, int lane) {
  const int n  = n0 + (lane & 15);
  const int kk = k0 + ((lane >> 4) << 4);
  const bf16* p = W + (size_t)n * ld + kk;
  v8bf lo = *(const v8bf*)p;
  v8bf hi = *(const v8bf*)(p + 8);
  v16bf b;
#pragma unroll
  for (int i = 0; i < 8; ++i) { b[i] = lo[i]; b[i + 8] = hi[i]; }
  return b;
}

// ---------------------------------------------------------------------------
// fp32 -> bf16 bulk convert (weight cache), n multiple of 256
__global__ void cvt_bf16_kernel(const float* __restrict__ src,
                                bf16* __restrict__ dst) {
  const int i = blockIdx.x * blockDim.x + threadIdx.x;
  dst[i] = (bf16)src[i];
}

// h init: fp32 copy + bf16 shadow of the recurrent state
__global__ void init_h_kernel(const float* __restrict__ h_in,
                              float* __restrict__ hF, bf16* __restrict__ hB) {
  const int i = blockIdx.x * blockDim.x + threadIdx.x;  // L_*B_*H_ threads
  const float v = h_in[i];
  hF[i] = v;
  hB[i] = (bf16)v;
}

// Embedding gather * sqrt(E) -> bf16, for all timesteps up front.
__global__ void embed_kernel(const int* __restrict__ toks,
                             const float* __restrict__ emb,
                             bf16* __restrict__ X0) {
  const int idx = blockIdx.x * blockDim.x + threadIdx.x;  // S_*B_*E_ threads
  const int e  = idx & (E_ - 1);
  const int sb = idx >> 9;  // /E_
  const int tok = toks[sb];
  X0[idx] = (bf16)(emb[(size_t)tok * E_ + e] * 22.62741699796952f);  // sqrt(512)
}

// ---------------------------------------------------------------------------
// Phase 1 of a GRU layer: r = sig(xWr^T+br + hUr^T), z = sig(xWz^T+bz + hUz^T),
// xwh = xWh^T + bh. Emits rh=bf16(r*h), z (f32), xwh (f32).
// grid 16 (H/32 col blocks), block 128 (4 waves; wave w owns rows 16w..16w+15,
// and computes BOTH 16-col tiles so A-fragments are reused across 10 WMMAs/K).
template <typename WT>
__global__ __launch_bounds__(128) void gru_rz_kernel(
    const bf16* __restrict__ x_bf, const bf16* __restrict__ h_bf,
    const float* __restrict__ h_f32,
    const WT* __restrict__ Wr, const WT* __restrict__ Ur,
    const WT* __restrict__ Wz, const WT* __restrict__ Uz,
    const WT* __restrict__ Wh,
    const float* __restrict__ br, const float* __restrict__ bz,
    const float* __restrict__ bh,
    bf16* __restrict__ rh_bf, float* __restrict__ z_f32,
    float* __restrict__ xwh) {
  const int lane = threadIdx.x & 31;
  const int wave = threadIdx.x >> 5;
  const int m0 = wave * 16;
  const int nb = blockIdx.x * 32;
  v8f ar[2] = {zero8(), zero8()};
  v8f az[2] = {zero8(), zero8()};
  v8f ah[2] = {zero8(), zero8()};
  for (int k = 0; k < H_; k += 32) {
    v16bf ax  = load_a(x_bf, H_, m0, k, lane);
    v16bf ahd = load_a(h_bf, H_, m0, k, lane);
#pragma unroll
    for (int nt = 0; nt < 2; ++nt) {
      const int n0 = nb + nt * 16;
      v16bf bm;
      bm = load_w16(Wr, H_, n0, k, lane); ar[nt] = wmma_bf16(ax,  bm, ar[nt]);
      bm = load_w16(Ur, H_, n0, k, lane); ar[nt] = wmma_bf16(ahd, bm, ar[nt]);
      bm = load_w16(Wz, H_, n0, k, lane); az[nt] = wmma_bf16(ax,  bm, az[nt]);
      bm = load_w16(Uz, H_, n0, k, lane); az[nt] = wmma_bf16(ahd, bm, az[nt]);
      bm = load_w16(Wh, H_, n0, k, lane); ah[nt] = wmma_bf16(ax,  bm, ah[nt]);
    }
  }
#pragma unroll
  for (int nt = 0; nt < 2; ++nt) {
    const int n  = nb + nt * 16 + (lane & 15);
    const int mb = m0 + ((lane >> 4) << 3);
    const float brv = br[n], bzv = bz[n], bhv = bh[n];
#pragma unroll
    for (int i = 0; i < 8; ++i) {
      const int m = mb + i;
      const float rv   = sigm(ar[nt][i] + brv);
      const float zv   = sigm(az[nt][i] + bzv);
      const float hold = h_f32[m * H_ + n];
      z_f32[m * H_ + n] = zv;
      xwh [m * H_ + n] = ah[nt][i] + bhv;
      rh_bf[m * H_ + n] = (bf16)(rv * hold);
    }
  }
}

// Phase 2: hc = tanh(xwh + (r*h)Uh^T); h = (1-z)h + z*hc.
// Writes fp32 + bf16 state; optionally stashes layer-1 output (A for logits)
// and the final hidden into d_out.
template <typename WT>
__global__ __launch_bounds__(128) void gru_upd_kernel(
    const bf16* __restrict__ rh_bf, const WT* __restrict__ Uh,
    const float* __restrict__ z_f32, const float* __restrict__ xwh,
    float* __restrict__ h_f32, bf16* __restrict__ h_bf,
    bf16* __restrict__ h2_dst, float* __restrict__ hout_dst) {
  const int lane = threadIdx.x & 31;
  const int wave = threadIdx.x >> 5;
  const int m0 = wave * 16;
  const int nb = blockIdx.x * 32;
  v8f acc[2] = {zero8(), zero8()};
  for (int k = 0; k < H_; k += 32) {
    v16bf a = load_a(rh_bf, H_, m0, k, lane);
#pragma unroll
    for (int nt = 0; nt < 2; ++nt) {
      v16bf b = load_w16(Uh, H_, nb + nt * 16, k, lane);
      acc[nt] = wmma_bf16(a, b, acc[nt]);
    }
  }
#pragma unroll
  for (int nt = 0; nt < 2; ++nt) {
    const int n  = nb + nt * 16 + (lane & 15);
    const int mb = m0 + ((lane >> 4) << 3);
#pragma unroll
    for (int i = 0; i < 8; ++i) {
      const int m = mb + i;
      const float hc   = tanhf(acc[nt][i] + xwh[m * H_ + n]);
      const float zv   = z_f32[m * H_ + n];
      const float hold = h_f32[m * H_ + n];
      const float hnew = (1.0f - zv) * hold + zv * hc;
      h_f32[m * H_ + n] = hnew;
      h_bf [m * H_ + n] = (bf16)hnew;
      if (h2_dst)   h2_dst[m * H_ + n] = (bf16)hnew;
      if (hout_dst) hout_dst[m * H_ + n] = hnew;
    }
  }
}

// ---------------------------------------------------------------------------
// logits[4096 x 32000] = H2[4096 x 512](bf16) @ fc_w[32000 x 512]^T + fc_b.
// grid (V/128, M/128), block 256 (8 waves): wave -> 32 rows x 64 cols
// (2 m-tiles x 4 n-tiles, B-fragments reused across both m-tiles).
template <typename WT>
__global__ __launch_bounds__(256) void logits_kernel(
    const bf16* __restrict__ A, const WT* __restrict__ fcw,
    const float* __restrict__ fcb, float* __restrict__ out) {
  const int lane = threadIdx.x & 31;
  const int wave = threadIdx.x >> 5;
  const int m0  = blockIdx.y * 128 + ((wave >> 1) << 5);  // 32-row strip
  const int n0b = blockIdx.x * 128 + ((wave & 1) << 6);   // 64-col strip
  v8f acc[2][4] = {{zero8(), zero8(), zero8(), zero8()},
                   {zero8(), zero8(), zero8(), zero8()}};
  for (int k = 0; k < H_; k += 32) {
    v16bf a0 = load_a(A, H_, m0,      k, lane);
    v16bf a1 = load_a(A, H_, m0 + 16, k, lane);
#pragma unroll
    for (int j = 0; j < 4; ++j) {
      v16bf b = load_w16(fcw, H_, n0b + 16 * j, k, lane);
      acc[0][j] = wmma_bf16(a0, b, acc[0][j]);
      acc[1][j] = wmma_bf16(a1, b, acc[1][j]);
    }
  }
#pragma unroll
  for (int mt = 0; mt < 2; ++mt) {
    const int mb = m0 + 16 * mt + ((lane >> 4) << 3);
#pragma unroll
    for (int j = 0; j < 4; ++j) {
      const int n = n0b + 16 * j + (lane & 15);
      const float bias = fcb[n];
#pragma unroll
      for (int i = 0; i < 8; ++i) {
        const size_t m = (size_t)(mb + i);
        out[m * V_ + n] = acc[mt][j][i] + bias;
      }
    }
  }
}

// ---------------------------------------------------------------------------
extern "C" void kernel_launch(void* const* d_in, const int* in_sizes, int n_in,
                              void* d_out, int out_size, void* d_ws, size_t ws_size,
                              hipStream_t stream) {
  (void)in_sizes; (void)n_in; (void)out_size;
  const int*   inputs = (const int*)d_in[0];    // (S,B) int32
  const float* hidden = (const float*)d_in[1];  // (L,B,H)
  const float* emb    = (const float*)d_in[2];  // (V,E)
  const float* Wr_w   = (const float*)d_in[3];
  const float* Wr_b   = (const float*)d_in[4];
  const float* Ur_w   = (const float*)d_in[5];
  const float* Wz_w   = (const float*)d_in[6];
  const float* Wz_b   = (const float*)d_in[7];
  const float* Uz_w   = (const float*)d_in[8];
  const float* Wh_w   = (const float*)d_in[9];
  const float* Wh_b   = (const float*)d_in[10];
  const float* Uh_w   = (const float*)d_in[11];
  const float* fc_w   = (const float*)d_in[12];
  const float* fc_b   = (const float*)d_in[13];
  float* out = (float*)d_out;

  // Workspace carve-up. Base ~8.7 MB; optional bf16 weight cache +39.1 MB.
  char* p = (char*)d_ws;
  bf16*  X0  = (bf16*)p;  p += (size_t)S_ * B_ * E_ * 2;  // embedded input (bf16)
  bf16*  H2  = (bf16*)p;  p += (size_t)S_ * B_ * H_ * 2;  // layer-1 outputs (bf16)
  float* hF  = (float*)p; p += (size_t)L_ * B_ * H_ * 4;  // recurrent state fp32
  bf16*  hB  = (bf16*)p;  p += (size_t)L_ * B_ * H_ * 2;  // recurrent state bf16
  bf16*  RH  = (bf16*)p;  p += (size_t)B_ * H_ * 2;       // r*h
  float* Z   = (float*)p; p += (size_t)B_ * H_ * 4;       // z
  float* XWH = (float*)p; p += (size_t)B_ * H_ * 4;       // x*Wh^T + bh

  const size_t gsz = (size_t)L_ * H_ * H_;                // gate matrix elems
  bf16* WrB = (bf16*)p; p += gsz * 2;
  bf16* UrB = (bf16*)p; p += gsz * 2;
  bf16* WzB = (bf16*)p; p += gsz * 2;
  bf16* UzB = (bf16*)p; p += gsz * 2;
  bf16* WhB = (bf16*)p; p += gsz * 2;
  bf16* UhB = (bf16*)p; p += gsz * 2;
  bf16* FCB = (bf16*)p; p += (size_t)V_ * H_ * 2;
  const bool cache = ((size_t)(p - (char*)d_ws) <= ws_size);

  init_h_kernel<<<(L_ * B_ * H_) / 256, 256, 0, stream>>>(hidden, hF, hB);
  embed_kernel<<<(S_ * B_ * E_) / 256, 256, 0, stream>>>(inputs, emb, X0);

  if (cache) {
    cvt_bf16_kernel<<<gsz / 256, 256, 0, stream>>>(Wr_w, WrB);
    cvt_bf16_kernel<<<gsz / 256, 256, 0, stream>>>(Ur_w, UrB);
    cvt_bf16_kernel<<<gsz / 256, 256, 0, stream>>>(Wz_w, WzB);
    cvt_bf16_kernel<<<gsz / 256, 256, 0, stream>>>(Uz_w, UzB);
    cvt_bf16_kernel<<<gsz / 256, 256, 0, stream>>>(Wh_w, WhB);
    cvt_bf16_kernel<<<gsz / 256, 256, 0, stream>>>(Uh_w, UhB);
    cvt_bf16_kernel<<<((size_t)V_ * H_) / 256, 256, 0, stream>>>(fc_w, FCB);
  }

  const size_t logits_elems = (size_t)S_ * B_ * V_;
  for (int t = 0; t < S_; ++t) {
    for (int l = 0; l < L_; ++l) {
      const bf16* x = (l == 0) ? (X0 + (size_t)t * B_ * E_) : hB;  // hB[0] = layer-0 out
      const int wo = l * H_ * H_;
      const int bo = l * H_;
      bf16*  h2d = (l == 1) ? (H2 + (size_t)t * B_ * H_) : (bf16*)nullptr;
      float* hod = (t == S_ - 1) ? (out + logits_elems + (size_t)l * B_ * H_)
                                 : (float*)nullptr;
      if (cache) {
        gru_rz_kernel<bf16><<<H_ / 32, 128, 0, stream>>>(
            x, hB + (size_t)l * B_ * H_, hF + (size_t)l * B_ * H_,
            WrB + wo, UrB + wo, WzB + wo, UzB + wo, WhB + wo,
            Wr_b + bo, Wz_b + bo, Wh_b + bo, RH, Z, XWH);
        gru_upd_kernel<bf16><<<H_ / 32, 128, 0, stream>>>(
            RH, UhB + wo, Z, XWH,
            hF + (size_t)l * B_ * H_, hB + (size_t)l * B_ * H_, h2d, hod);
      } else {
        gru_rz_kernel<float><<<H_ / 32, 128, 0, stream>>>(
            x, hB + (size_t)l * B_ * H_, hF + (size_t)l * B_ * H_,
            Wr_w + wo, Ur_w + wo, Wz_w + wo, Uz_w + wo, Wh_w + wo,
            Wr_b + bo, Wz_b + bo, Wh_b + bo, RH, Z, XWH);
        gru_upd_kernel<float><<<H_ / 32, 128, 0, stream>>>(
            RH, Uh_w + wo, Z, XWH,
            hF + (size_t)l * B_ * H_, hB + (size_t)l * B_ * H_, h2d, hod);
      }
    }
  }

  if (cache) {
    logits_kernel<bf16><<<dim3(V_ / 128, (S_ * B_) / 128), 256, 0, stream>>>(
        H2, FCB, fc_b, out);
  } else {
    logits_kernel<float><<<dim3(V_ / 128, (S_ * B_) / 128), 256, 0, stream>>>(
        H2, fc_w, fc_b, out);
  }
}
